// SelfAttention_90580860273105
// MI455X (gfx1250) — matrix operand order repaired and checked by reading, not compiled, not run on hardware
//
#include <hip/hip_runtime.h>

typedef __attribute__((ext_vector_type(16))) _Float16 v16h;
typedef __attribute__((ext_vector_type(8)))  _Float16 v8h;
typedef __attribute__((ext_vector_type(8)))  float    v8f;

#define B_ 8
#define N_ 2048
#define C_ 256
#define M_ (B_*N_)   // 16384 tokens total

// ---------------- WMMA fragment loaders (CDNA5 16x16x32 f16 layouts) ----------------
// A (16x32, MxK) row-major source, leading dim ld (elements):
//   lane L: row = L&15 ; kb = 8*(L>>4) ; elems 0..7 -> K=kb..kb+7 ; elems 8..15 -> K=kb+16..kb+23
__device__ __forceinline__ v16h load_a16(const _Float16* __restrict__ tile, int ld, int lane) {
    int row = lane & 15;
    int kb  = (lane >> 4) << 3;
    const _Float16* p = tile + row * ld + kb;
    v8h lo = *(const v8h*)(p);
    v8h hi = *(const v8h*)(p + 16);
    return __builtin_shufflevector(lo, hi, 0,1,2,3,4,5,6,7,8,9,10,11,12,13,14,15);
}

// B (32x16, KxN) supplied as "B^T" (row = output column, contiguous in K), leading dim ld:
//   lane L: col = L&15 ; kb = 8*(L>>4) ; elems 0..7 -> K=kb..kb+7 ; elems 8..15 -> K=kb+16..kb+23
__device__ __forceinline__ v16h load_bT16(const _Float16* __restrict__ tileT, int ld, int lane) {
    int col  = lane & 15;
    int koff = (lane >> 4) << 3;
    const _Float16* p = tileT + col * ld + koff;
    v8h lo = *(const v8h*)(p);
    v8h hi = *(const v8h*)(p + 16);
    return __builtin_shufflevector(lo, hi, 0,1,2,3,4,5,6,7,8,9,10,11,12,13,14,15);
}

// ---------------- conversion kernels ----------------
__global__ void k_cvt_x(const float* __restrict__ x, _Float16* __restrict__ xh, int n) {
    int i = blockIdx.x * blockDim.x + threadIdx.x;
    int stride = gridDim.x * blockDim.x;
    for (; i < n; i += stride) xh[i] = (_Float16)x[i];
}

// Wt[w][f][c] = W_w[c][f]  (transpose so the B operand is K-contiguous per column)
__global__ void k_cvt_w(const float* __restrict__ Wq, const float* __restrict__ Wk,
                        const float* __restrict__ Wv, _Float16* __restrict__ Wt) {
    int i = blockIdx.x * blockDim.x + threadIdx.x;
    if (i >= 3 * C_ * C_) return;
    int w = i / (C_ * C_);
    int r = i % (C_ * C_);
    int c = r / C_, f = r % C_;
    const float* W = (w == 0) ? Wq : (w == 1) ? Wk : Wv;
    Wt[w * C_ * C_ + f * C_ + c] = (_Float16)W[c * C_ + f];
}

// ---------------- QKV projection: [16384 x 256] x [256 x 256] (x3 weights) ----------------
// Q,K stored row-major f16 [M, C]; V stored transposed per batch: Vt[b][f][n]
__global__ void k_proj(const _Float16* __restrict__ xh, const _Float16* __restrict__ Wt,
                       const float* __restrict__ bq, const float* __restrict__ bk,
                       const float* __restrict__ bv,
                       _Float16* __restrict__ Qh, _Float16* __restrict__ Kh,
                       _Float16* __restrict__ Vt) {
    int wave = (blockIdx.x * blockDim.x + threadIdx.x) >> 5;
    int lane = threadIdx.x & 31;
    const int tilesM = M_ / 16, tilesN = C_ / 16;      // 1024, 16
    int w  = wave / (tilesM * tilesN);
    int r  = wave % (tilesM * tilesN);
    int tm = r / tilesN, tn = r % tilesN;

    const _Float16* Atile = xh + (size_t)tm * 16 * C_;
    const _Float16* Btile = Wt + (size_t)w * C_ * C_ + (size_t)tn * 16 * C_;
    v8f acc = {};
#pragma unroll
    for (int ks = 0; ks < C_ / 32; ++ks) {
        v16h a = load_a16(Atile + ks * 32, C_, lane);
        v16h b = load_bT16(Btile + ks * 32, C_, lane);
        acc = __builtin_amdgcn_wmma_f32_16x16x32_f16(false, a, false, b, (short)0, acc, false, false);
    }
    const float* bias = (w == 0) ? bq : (w == 1) ? bk : bv;
    int col   = lane & 15;
    int rbase = (lane >> 4) << 3;
    int f     = tn * 16 + col;
    float bb  = bias[f];
    if (w < 2) {
        _Float16* dst = (w == 0) ? Qh : Kh;
#pragma unroll
        for (int i = 0; i < 8; ++i) {
            int t = tm * 16 + rbase + i;
            dst[(size_t)t * C_ + f] = (_Float16)(acc[i] + bb);
        }
    } else {
#pragma unroll
        for (int i = 0; i < 8; ++i) {
            int t  = tm * 16 + rbase + i;
            int bi = t >> 11;          // / 2048
            int n  = t & (N_ - 1);
            Vt[((size_t)bi * C_ + f) * N_ + n] = (_Float16)(acc[i] + bb);
        }
    }
}

// ---------------- fused scores + softmax ----------------
// One 256-thread block handles a 16-row strip of S[b] = Q[b]K[b]^T (16 x 2048) entirely in
// LDS (128 KB of the WGP's 320 KB), then does the row softmax and writes attn exactly once.
// Wave w computes column tiles tn = w, w+8, ..., w+120; the 8 Q A-fragments of the strip are
// loaded once into registers and reused for all 16 tiles.
__global__ void k_scores_softmax(const _Float16* __restrict__ Qh, const _Float16* __restrict__ Kh,
                                 float* __restrict__ attn) {
    extern __shared__ char smem[];
    float* S      = (float*)smem;                       // 16 * 2048 f32 = 128 KB
    float* redbuf = (float*)(smem + 16 * N_ * 4);       // 256 f32
    float* rowred = redbuf + 256;                       // 16 f32

    int lane = threadIdx.x & 31;
    int w    = threadIdx.x >> 5;                        // wave 0..7
    int b    = blockIdx.x >> 7;                         // / 128 strips
    int strip= blockIdx.x & 127;
    int m0   = strip * 16;

    // ---- phase 1: WMMA scores into LDS ----
    const _Float16* Atile = Qh + ((size_t)b * N_ + m0) * C_;
    v16h afr[8];
#pragma unroll
    for (int ks = 0; ks < 8; ++ks) afr[ks] = load_a16(Atile + ks * 32, C_, lane);

    int col = lane & 15, rbase = (lane >> 4) << 3;
#pragma unroll 1
    for (int t = 0; t < 16; ++t) {
        int tn = w + t * 8;
        const _Float16* Btile = Kh + ((size_t)b * N_ + tn * 16) * C_;
        v8f acc = {};
#pragma unroll
        for (int ks = 0; ks < 8; ++ks) {
            v16h bf = load_bT16(Btile + ks * 32, C_, lane);
            acc = __builtin_amdgcn_wmma_f32_16x16x32_f16(false, afr[ks], false, bf, (short)0, acc, false, false);
        }
#pragma unroll
        for (int i = 0; i < 8; ++i)
            S[(rbase + i) * N_ + tn * 16 + col] = acc[i];
    }
    __syncthreads();

    // ---- phase 2: row softmax in LDS (16 threads per row, 128 elems each) ----
    int r   = threadIdx.x >> 4;                         // row 0..15
    int sub = threadIdx.x & 15;
    float* Sr = S + r * N_;

    float mx = -3.4e38f;
#pragma unroll 8
    for (int j = 0; j < 128; ++j) mx = fmaxf(mx, Sr[sub + j * 16]);
    redbuf[threadIdx.x] = mx;
    __syncthreads();
    if (sub == 0) {
        float mm = redbuf[r * 16];
        for (int t = 1; t < 16; ++t) mm = fmaxf(mm, redbuf[r * 16 + t]);
        rowred[r] = mm;
    }
    __syncthreads();
    mx = rowred[r];

    float sum = 0.f;
#pragma unroll 8
    for (int j = 0; j < 128; ++j) {
        float e = __expf(Sr[sub + j * 16] - mx);
        Sr[sub + j * 16] = e;
        sum += e;
    }
    redbuf[threadIdx.x] = sum;
    __syncthreads();
    if (sub == 0) {
        float ss = redbuf[r * 16];
        for (int t = 1; t < 16; ++t) ss += redbuf[r * 16 + t];
        rowred[r] = ss;
    }
    __syncthreads();
    float inv = 1.0f / rowred[r];

    float* dst = attn + ((size_t)b * N_ + m0 + r) * N_;
#pragma unroll 8
    for (int j = 0; j < 128; ++j) dst[sub + j * 16] = Sr[sub + j * 16] * inv;
}

// ---------------- out = x + gamma * mask^T ⊙ (A^T @ V) ----------------
// One wave register-blocks a full 16-token x 256-feature output strip (16 x v8f accumulators),
// so each attn element is gathered exactly once. The 8 waves of a block share a 32x256 Vt
// k-slab, double-buffered in LDS and filled with GLOBAL_LOAD_ASYNC_TO_LDS_B128 (ASYNCcnt)
// so the next slab streams in while the current one feeds 16 WMMAs per k-step.
// Row stride padded to 40 halfwords for conflict-free B-frag reads.
#define VS_LD 40
__global__ void k_out(const float* __restrict__ attn, const _Float16* __restrict__ Vt,
                      const float* __restrict__ x, const float* __restrict__ mask,
                      const float* __restrict__ gamma, float* __restrict__ out) {
    __shared__ _Float16 vs[2 * 256 * VS_LD];            // 2 x 20 KB slabs

    int lane = threadIdx.x & 31;
    int w    = threadIdx.x >> 5;
    int b    = blockIdx.x >> 4;                         // / 16 groups
    int tm   = (blockIdx.x & 15) * 8 + w;               // token tile 0..127
    int m0   = tm * 16;

    const float*    A  = attn + (size_t)b * N_ * N_;
    const _Float16* Vb = Vt + (size_t)b * C_ * N_;
    int row = lane & 15;
    int kb  = (lane >> 4) << 3;
    int koff= (lane >> 4) << 3;

    // LDS aperture lives in addr[63:32] (ISA 10.2), so the low 32 bits of a generic
    // shared pointer are the workgroup-relative LDS byte offset the async VDST wants.
    unsigned lds0 = (unsigned)(uintptr_t)(vs + threadIdx.x * VS_LD);
    unsigned lds1 = (unsigned)(uintptr_t)(vs + 256 * VS_LD + threadIdx.x * VS_LD);
    const _Float16* srow = Vb + (size_t)threadIdx.x * N_;   // this thread's feature row

    // async copy of one 64B row chunk: IOFFSET applies to BOTH global and LDS address
#define ASYNC_ROW(LDS, GPTR)                                                    \
    asm volatile("global_load_async_to_lds_b128 %0, %1, off\n\t"                \
                 "global_load_async_to_lds_b128 %0, %1, off offset:16\n\t"      \
                 "global_load_async_to_lds_b128 %0, %1, off offset:32\n\t"      \
                 "global_load_async_to_lds_b128 %0, %1, off offset:48"          \
                 :: "v"(LDS), "v"(GPTR) : "memory")

    v8f zero = {};
    v8f acc[16];
#pragma unroll
    for (int tf = 0; tf < 16; ++tf) acc[tf] = zero;

    // prime buffer 0 with k0 = 0
    ASYNC_ROW(lds0, srow);

    for (int k0 = 0; k0 < N_; k0 += 32) {
        int cur = (k0 >> 5) & 1;
        // kick off the next slab into the other buffer, then wait for the current one.
        // ASYNCcnt completes in order: after issuing 4 more, "<= 4" means ours landed.
        if (k0 + 32 < N_) {
            if (cur == 0) { ASYNC_ROW(lds1, srow + k0 + 32); }
            else          { ASYNC_ROW(lds0, srow + k0 + 32); }
            asm volatile("s_wait_asynccnt 0x4" ::: "memory");
        } else {
            asm volatile("s_wait_asynccnt 0x0" ::: "memory");
        }
        __syncthreads();

        const _Float16* slab = vs + cur * 256 * VS_LD;

        // A^T fragment gathered from f32 attn (inline f16 convert):
        //   elems 0..7 -> n = k0+kb+e ; elems 8..15 -> n = k0+kb+16+(e-8)
        v16h a;
        const float* p0 = A + (size_t)(k0 + kb) * N_ + m0 + row;
        const float* p1 = p0 + (size_t)16 * N_;
#pragma unroll
        for (int e = 0; e < 8; ++e) a[e]     = (_Float16)p0[(size_t)e * N_];
#pragma unroll
        for (int e = 0; e < 8; ++e) a[8 + e] = (_Float16)p1[(size_t)e * N_];

        // 16 feature tiles from the shared LDS slab
#pragma unroll
        for (int tf = 0; tf < 16; ++tf) {
            int colf = tf * 16 + (lane & 15);
            const _Float16* p = slab + colf * VS_LD + koff;
            v8h lo = *(const v8h*)(p);
            v8h hi = *(const v8h*)(p + 16);
            v16h bf = __builtin_shufflevector(lo, hi, 0,1,2,3,4,5,6,7,8,9,10,11,12,13,14,15);
            acc[tf] = __builtin_amdgcn_wmma_f32_16x16x32_f16(false, a, false, bf, (short)0, acc[tf], false, false);
        }
        __syncthreads();   // all waves done reading before this buffer is re-filled
    }
#undef ASYNC_ROW

    float g = gamma[0];
    int col = lane & 15, rbase = (lane >> 4) << 3;
#pragma unroll
    for (int tf = 0; tf < 16; ++tf) {
        int f = tf * 16 + col;
#pragma unroll
        for (int i = 0; i < 8; ++i) {
            int m = m0 + rbase + i;
            size_t xi = ((size_t)b * N_ + m) * C_ + f;
            float mk = mask[((size_t)b * C_ + f) * N_ + m];
            out[xi] = x[xi] + g * mk * acc[tf][i];
        }
    }
}

// ---------------- launcher ----------------
extern "C" void kernel_launch(void* const* d_in, const int* in_sizes, int n_in,
                              void* d_out, int out_size, void* d_ws, size_t ws_size,
                              hipStream_t stream) {
    (void)in_sizes; (void)n_in; (void)out_size; (void)ws_size;
    const float* x     = (const float*)d_in[0];
    const float* mask  = (const float*)d_in[1];
    const float* Wq    = (const float*)d_in[2];
    const float* bq    = (const float*)d_in[3];
    const float* Wk    = (const float*)d_in[4];
    const float* bk    = (const float*)d_in[5];
    const float* Wv    = (const float*)d_in[6];
    const float* bv    = (const float*)d_in[7];
    const float* gamma = (const float*)d_in[8];

    float* out  = (float*)d_out;                       // [B,N,C]
    float* attn = out + (size_t)B_ * N_ * C_;          // [B,N,N]

    _Float16* xh = (_Float16*)d_ws;                    // 16384*256 f16
    _Float16* Qh = xh + (size_t)M_ * C_;
    _Float16* Kh = Qh + (size_t)M_ * C_;
    _Float16* Vt = Kh + (size_t)M_ * C_;               // per-batch transposed [B][C][N]
    _Float16* Wt = Vt + (size_t)M_ * C_;               // 3*[C][C] transposed

    k_cvt_x<<<2048, 256, 0, stream>>>(x, xh, M_ * C_);
    k_cvt_w<<<(3 * C_ * C_ + 255) / 256, 256, 0, stream>>>(Wq, Wk, Wv, Wt);
    // 3 * 1024 * 16 = 49152 waves, 8 waves/block
    k_proj<<<6144, 256, 0, stream>>>(xh, Wt, bq, bk, bv, Qh, Kh, Vt);
    // 8 batches * 128 strips; 128 KB scores strip + 1.1 KB reductions of dynamic LDS
    size_t smem = (size_t)16 * N_ * 4 + (256 + 16) * 4;
    k_scores_softmax<<<1024, 256, smem, stream>>>(Qh, Kh, attn);
    // 8 batches * 16 groups of 8 token tiles
    k_out<<<128, 256, 0, stream>>>(attn, Vt, x, mask, gamma, out);
}
